// DeformableSpatialAttention_84035330113696
// MI455X (gfx1250) — matrix operand order repaired and checked by reading, not compiled, not run on hardware
//
#include <hip/hip_runtime.h>
#include <hip/hip_bf16.h>
#include <math.h>

// ---------------------------------------------------------------------------
// DeformableSpatialAttention for MI455X (gfx1250), wave32 + WMMA bf16
// + async global->LDS prefetch of the V strip (ASYNCcnt path).
// B=4, S=1024, E=256, H=8, D=32.
// ---------------------------------------------------------------------------

#define BB 4
#define SS 1024
#define EE 256
#define HH 8
#define DD 32

typedef __attribute__((ext_vector_type(16))) __bf16 v16bf;
typedef __attribute__((ext_vector_type(8)))  __bf16 v8bf;
typedef __attribute__((ext_vector_type(8)))  float  v8f;

// ---------------------------------------------------------------------------
// f32 -> bf16 elementwise convert
// ---------------------------------------------------------------------------
__global__ __launch_bounds__(256) void f32_to_bf16_kernel(const float* __restrict__ in,
                                                          __bf16* __restrict__ out, int n) {
    int i = blockIdx.x * blockDim.x + threadIdx.x;
    if (i < n) out[i] = (__bf16)in[i];
}

// ---------------------------------------------------------------------------
// Offset MLP: off = gelu(x @ W1 + b1) @ W2 + b2   (exact erf GELU)
// One thread per token (4096 tokens). ~2% of total FLOPs -> plain VALU.
// ---------------------------------------------------------------------------
__global__ __launch_bounds__(256) void offset_mlp_kernel(const float* __restrict__ x,
                                                         const float* __restrict__ W1,
                                                         const float* __restrict__ b1,
                                                         const float* __restrict__ W2,
                                                         const float* __restrict__ b2,
                                                         float* __restrict__ offs) {
    int t = blockIdx.x * blockDim.x + threadIdx.x;
    if (t >= BB * SS) return;
    const float* xr = x + (size_t)t * EE;
    float oacc[2 * HH];
#pragma unroll
    for (int o = 0; o < 2 * HH; ++o) oacc[o] = b2[o];
    for (int hc = 0; hc < EE / 2; ++hc) {
        float d = b1[hc];
        for (int k = 0; k < EE; ++k) d += xr[k] * W1[k * (EE / 2) + hc];
        float g = 0.5f * d * (1.0f + erff(d * 0.70710678118654752f));
#pragma unroll
        for (int o = 0; o < 2 * HH; ++o) oacc[o] += g * W2[hc * (2 * HH) + o];
    }
#pragma unroll
    for (int o = 0; o < 2 * HH; ++o) offs[(size_t)t * (2 * HH) + o] = oacc[o];
}

// ---------------------------------------------------------------------------
// WMMA bf16 GEMM: C(4096x256) = A(4096x256) @ Bw(256x256)
// mode 0: bf16 out row-major, *scale (Q, scale = 1/sqrt(D) folded in)
// mode 1: bf16 out transposed to [b][h][d][S]   (K^T for attention B operand)
// mode 2: bf16 out row-major                    (V)
// mode 3: f32 out row-major + bias              (final projection -> d_out)
// Block = 128 threads = 4 waves; wave computes one 16x16 tile, K-loop 8x32.
// ---------------------------------------------------------------------------
__global__ __launch_bounds__(128) void wmma_gemm_kernel(const __bf16* __restrict__ A,
                                                        const __bf16* __restrict__ Bw,
                                                        void* __restrict__ Cout,
                                                        const float* __restrict__ bias,
                                                        int mode, float scale) {
    const int lane = threadIdx.x & 31;
    const int wave = threadIdx.x >> 5;
    const int L15  = lane & 15;
    const int half = lane >> 4;
    const int m0 = blockIdx.x * 16;
    const int n0 = (blockIdx.y * 4 + wave) * 16;

    v8f acc = {};
    for (int k0 = 0; k0 < EE; k0 += 32) {
        // A operand: lane holds row m=L15; K chunks [k0+8*half .. +7] and [k0+16+8*half .. +7]
        const __bf16* ap = A + (size_t)(m0 + L15) * EE + k0 + half * 8;
        v16bf a;
        *(v8bf*)&a       = *(const v8bf*)ap;
        *((v8bf*)&a + 1) = *(const v8bf*)(ap + 16);
        // B operand: lane holds column n0+L15; K rows k0+8*half+e and k0+16+8*half+e
        v16bf b;
#pragma unroll
        for (int e = 0; e < 8; ++e) {
            b[e]     = Bw[(size_t)(k0 + half * 8 + e) * EE + n0 + L15];
            b[8 + e] = Bw[(size_t)(k0 + 16 + half * 8 + e) * EE + n0 + L15];
        }
        acc = __builtin_amdgcn_wmma_f32_16x16x32_bf16(false, a, false, b,
                                                      (short)0, acc, false, false);
    }

#pragma unroll
    for (int r = 0; r < 8; ++r) {
        int gm = m0 + r + 8 * half;   // D layout: VGPR r -> row r (lanes 0-15) / 8+r (lanes 16-31)
        int gn = n0 + L15;            // N = lane%16
        float v = acc[r] * scale;
        if (mode == 3) {
            ((float*)Cout)[(size_t)gm * EE + gn] = v + bias[gn];
        } else if (mode == 1) {
            int bb = gm >> 10, s = gm & (SS - 1);
            int h  = gn >> 5,  d = gn & (DD - 1);
            ((__bf16*)Cout)[((size_t)((bb * HH + h) * DD + d) << 10) + s] = (__bf16)v;
        } else {
            ((__bf16*)Cout)[(size_t)gm * EE + gn] = (__bf16)v;
        }
    }
}

// ---------------------------------------------------------------------------
// Async global->LDS copy (CDNA5 ASYNCcnt path), GV addressing mode.
// lds_off is a byte offset into this workgroup's LDS allocation.
// ---------------------------------------------------------------------------
__device__ inline void async_copy_b128(unsigned lds_off, const void* gptr) {
    asm volatile("global_load_async_to_lds_b128 %0, %1, off"
                 :: "v"(lds_off), "v"((unsigned long long)(uintptr_t)gptr)
                 : "memory");
}
__device__ inline void wait_asynccnt0() {
    asm volatile("s_wait_asynccnt 0x0" ::: "memory");
}

// ---------------------------------------------------------------------------
// Fused deformable attention, one (b,h,16-query-tile) per block, 2 waves.
// LDS: srow[16][LD] holds one gathered score row-set (y0 then y1, reused),
//      dbuf[16][LD] accumulates the bilinear blend -> attn -> AV A-operand,
//      vtile[1024 rows x 64B data + 16B pad] holds the async-prefetched V strip.
// ---------------------------------------------------------------------------
#define LD 1028        // 1024 + 4 pad: avoids LDS bank conflicts on WMMA D scatter
#define VROW 80        // 64B of V data + 16B pad -> stride-20-bank pattern, 2-way max

__device__ inline float wave_max32(float v) {
#pragma unroll
    for (int o = 16; o > 0; o >>= 1) v = fmaxf(v, __shfl_xor(v, o, 32));
    return v;
}
__device__ inline float wave_sum32(float v) {
#pragma unroll
    for (int o = 16; o > 0; o >>= 1) v += __shfl_xor(v, o, 32);
    return v;
}

__global__ __launch_bounds__(64) void deform_attn_kernel(const __bf16* __restrict__ qb,   // [B*S][E], scale pre-folded
                                                         const __bf16* __restrict__ kTb,  // [b][h][d][S]
                                                         const __bf16* __restrict__ vb,   // [B*S][E]
                                                         const float*  __restrict__ offs, // [B*S][H][2]
                                                         __bf16* __restrict__ ctxb) {     // [B*S][E]
    extern __shared__ float smem[];
    float* srow = smem;            // 16*LD floats
    float* dbuf = smem + 16 * LD;  // 16*LD floats
    const char* vtile = (const char*)(smem + 2 * 16 * LD);  // 1024 * VROW bytes

    __shared__ int   p_yc[2][16];
    __shared__ float p_w[2][16];   // row-set weight, zeroed when y-corner invalid
    __shared__ int   p_fx[16];
    __shared__ float p_wx[16];

    const int bid = blockIdx.x;
    const int qt = bid & 63;
    const int h  = (bid >> 6) & (HH - 1);
    const int b  = bid >> 9;
    const int i0 = qt * 16;

    const int tid  = threadIdx.x;
    const int lane = tid & 31;
    const int wave = tid >> 5;
    const int L15  = lane & 15;
    const int half = lane >> 4;

    // ---- async prefetch of this (b,h)'s V strip into LDS -----------------
    // Issued first; completes while the score/blend/softmax phases run.
    // Dynamic LDS begins right after static LDS allocations.
    const unsigned vt_lds = __builtin_amdgcn_groupstaticsize() + 2u * 16 * LD * 4;
    for (int c = tid; c < SS * 4; c += 64) {           // 4096 x b128 chunks
        int row = c >> 2, part = c & 3;                 // row = key j, part = 8-elem d chunk
        const __bf16* g = vb + (size_t)(b * SS + row) * EE + h * DD + part * 8;
        async_copy_b128(vt_lds + (unsigned)row * VROW + (unsigned)part * 16, g);
    }

    // ---- per-query bilinear parameters -----------------------------------
    if (tid < 16) {
        int i = i0 + tid;
        float ox = offs[((size_t)(b * SS + i) * HH + h) * 2 + 0];
        float oy = offs[((size_t)(b * SS + i) * HH + h) * 2 + 1];
        float ysf = (float)i + oy;
        float y0f = floorf(ysf);
        float wy  = ysf - y0f;
        int   y0  = (int)y0f;
        int   y1  = y0 + 1;
        p_yc[0][tid] = min(max(y0, 0), SS - 1);
        p_yc[1][tid] = min(max(y1, 0), SS - 1);
        p_w[0][tid]  = (y0 >= 0 && y0 <= SS - 1) ? (1.0f - wy) : 0.0f;
        p_w[1][tid]  = (y1 >= 0 && y1 <= SS - 1) ? wy : 0.0f;
        float fxf = floorf(ox);
        p_fx[tid] = (int)fxf;
        p_wx[tid] = ox - fxf;
    }
    __syncthreads();

    // ---- two row-sets: scores via WMMA -> LDS, then shifted blend --------
    for (int ps = 0; ps < 2; ++ps) {
        // gathered A: lane holds q row p_yc[ps][L15] of this (b,h)
        const __bf16* ap = qb + (size_t)(b * SS + p_yc[ps][L15]) * EE + h * DD + half * 8;
        v16bf a;
        *(v8bf*)&a       = *(const v8bf*)ap;
        *((v8bf*)&a + 1) = *(const v8bf*)(ap + 16);

        for (int jt = wave; jt < SS / 16; jt += 2) {
            const __bf16* bp = kTb + ((size_t)((b * HH + h) * DD) << 10) + jt * 16 + L15;
            if (jt + 2 < SS / 16) __builtin_prefetch(bp + 32, 0, 3);  // global_prefetch_b8
            v16bf bm;
#pragma unroll
            for (int e = 0; e < 8; ++e) {
                bm[e]     = bp[(size_t)(half * 8 + e) << 10];
                bm[8 + e] = bp[(size_t)(16 + half * 8 + e) << 10];
            }
            v8f c = {};
            c = __builtin_amdgcn_wmma_f32_16x16x32_bf16(false, a, false, bm,
                                                        (short)0, c, false, false);
#pragma unroll
            for (int r = 0; r < 8; ++r)
                srow[(r + 8 * half) * LD + jt * 16 + L15] = c[r];
        }
        __syncthreads();

        // constant-shift bilinear blend of this row-set into dbuf
        for (int m = 0; m < 16; ++m) {
            float w  = p_w[ps][m];
            float wx = p_wx[m];
            int   fx = p_fx[m];
            for (int j = tid; j < SS; j += 64) {
                int xa = j + fx;
                float sa = (xa >= 0     && xa < SS)     ? srow[m * LD + xa]     : 0.0f;
                float sb = (xa + 1 >= 0 && xa + 1 < SS) ? srow[m * LD + xa + 1] : 0.0f;
                float contrib = w * ((1.0f - wx) * sa + wx * sb);
                if (ps == 0) dbuf[m * LD + j] = contrib;
                else         dbuf[m * LD + j] += contrib;
            }
        }
        __syncthreads();
    }

    // ---- softmax over j (rows split across the 2 waves) ------------------
    for (int m = wave * 8; m < wave * 8 + 8; ++m) {
        float mx = -3.402823466e+38f;
        for (int j = lane; j < SS; j += 32) mx = fmaxf(mx, dbuf[m * LD + j]);
        mx = wave_max32(mx);
        float sum = 0.0f;
        for (int j = lane; j < SS; j += 32) {
            float e = __expf(dbuf[m * LD + j] - mx);
            dbuf[m * LD + j] = e;
            sum += e;
        }
        sum = wave_sum32(sum);
        float inv = 1.0f / sum;
        for (int j = lane; j < SS; j += 32) dbuf[m * LD + j] *= inv;
    }

    // V prefetch must be complete & visible to both waves before AV.
    wait_asynccnt0();
    __syncthreads();

    // ---- AV: out(16 x 32) = attn(16 x 1024) @ V(1024 x 32), WMMA ---------
    const int n0 = wave * 16;  // wave 0 -> d 0..15, wave 1 -> d 16..31
    v8f acc = {};
    for (int kt = 0; kt < SS / 32; ++kt) {
        // A operand from attn in LDS (fp32 -> bf16)
        v16bf a;
        const float* arow = dbuf + L15 * LD + kt * 32 + half * 8;
#pragma unroll
        for (int e = 0; e < 8; ++e) {
            a[e]     = (__bf16)arow[e];
            a[8 + e] = (__bf16)arow[16 + e];
        }
        // B operand from the async-staged V tile: lane = column d = n0+L15, elements over key rows
        const char* vrow = vtile + (size_t)(kt * 32) * VROW + (n0 + L15) * 2;
        v16bf bm;
#pragma unroll
        for (int e = 0; e < 8; ++e) {
            bm[e]     = *(const __bf16*)(vrow + (size_t)(half * 8 + e) * VROW);
            bm[8 + e] = *(const __bf16*)(vrow + (size_t)(16 + half * 8 + e) * VROW);
        }
        acc = __builtin_amdgcn_wmma_f32_16x16x32_bf16(false, a, false, bm,
                                                      (short)0, acc, false, false);
    }
#pragma unroll
    for (int r = 0; r < 8; ++r) {
        int i = i0 + r + 8 * half;
        ctxb[(size_t)(b * SS + i) * EE + h * DD + n0 + L15] = (__bf16)acc[r];
    }
}

// ---------------------------------------------------------------------------
// Host-side orchestration (all launches on `stream`, graph-capture safe)
// ---------------------------------------------------------------------------
extern "C" void kernel_launch(void* const* d_in, const int* in_sizes, int n_in,
                              void* d_out, int out_size, void* d_ws, size_t ws_size,
                              hipStream_t stream) {
    (void)in_sizes; (void)n_in; (void)out_size;
    const float* x      = (const float*)d_in[0];
    const float* Wq     = (const float*)d_in[1];
    const float* Wk     = (const float*)d_in[2];
    const float* Wv     = (const float*)d_in[3];
    const float* W_off1 = (const float*)d_in[4];
    const float* b_off1 = (const float*)d_in[5];
    const float* W_off2 = (const float*)d_in[6];
    const float* b_off2 = (const float*)d_in[7];
    const float* W_out  = (const float*)d_in[8];
    const float* b_out  = (const float*)d_in[9];

    const size_t MB = 1u << 20;
    const size_t WS_NEEDED = 10 * MB + 4 * (EE * EE * 2) + (size_t)BB * SS * 2 * HH * 4;
    if (ws_size < WS_NEEDED) return;

    char* ws = (char*)d_ws;
    __bf16* xb   = (__bf16*)(ws + 0 * MB);
    __bf16* qb   = (__bf16*)(ws + 2 * MB);
    __bf16* kTb  = (__bf16*)(ws + 4 * MB);
    __bf16* vb   = (__bf16*)(ws + 6 * MB);
    __bf16* ctxb = (__bf16*)(ws + 8 * MB);
    __bf16* wqb  = (__bf16*)(ws + 10 * MB);
    __bf16* wkb  = wqb + EE * EE;
    __bf16* wvb  = wkb + EE * EE;
    __bf16* wob  = wvb + EE * EE;
    float*  offs = (float*)((char*)(wob + EE * EE));

    const int nTok = BB * SS * EE;          // 1,048,576
    const int nW   = EE * EE;               // 65,536
    f32_to_bf16_kernel<<<(nTok + 255) / 256, 256, 0, stream>>>(x,     xb,  nTok);
    f32_to_bf16_kernel<<<(nW + 255) / 256,   256, 0, stream>>>(Wq,    wqb, nW);
    f32_to_bf16_kernel<<<(nW + 255) / 256,   256, 0, stream>>>(Wk,    wkb, nW);
    f32_to_bf16_kernel<<<(nW + 255) / 256,   256, 0, stream>>>(Wv,    wvb, nW);
    f32_to_bf16_kernel<<<(nW + 255) / 256,   256, 0, stream>>>(W_out, wob, nW);

    offset_mlp_kernel<<<(BB * SS + 255) / 256, 256, 0, stream>>>(x, W_off1, b_off1,
                                                                 W_off2, b_off2, offs);

    const float scale = 0.17677669529663689f;  // 1/sqrt(D), folded into Q
    dim3 gg(BB * SS / 16, EE / 16 / 4);
    wmma_gemm_kernel<<<gg, 128, 0, stream>>>(xb, wqb, (void*)qb,  nullptr, 0, scale);
    wmma_gemm_kernel<<<gg, 128, 0, stream>>>(xb, wkb, (void*)kTb, nullptr, 1, 1.0f);
    wmma_gemm_kernel<<<gg, 128, 0, stream>>>(xb, wvb, (void*)vb,  nullptr, 2, 1.0f);

    // srow + dbuf (fp32) + padded V tile; 208.5 KB, within the 320 KB WGP LDS.
    const size_t attn_lds = (size_t)2 * 16 * LD * sizeof(float) + (size_t)SS * VROW;
    deform_attn_kernel<<<BB * HH * (SS / 16), 64, attn_lds, stream>>>(qb, kTb, vb, offs, ctxb);

    wmma_gemm_kernel<<<gg, 128, 0, stream>>>(ctxb, wob, d_out, b_out, 3, 1.0f);
}
